// DownBlock_res_dct1_15315853378229
// MI455X (gfx1250) — matrix operand order repaired and checked by reading, compiled, not run on hardware
//
#include <hip/hip_runtime.h>
#include <hip/hip_bf16.h>
#include <stdint.h>

// MI455X (gfx1250) implementation.
// Pipeline: maxpool+DCT+IDCT-remap (k1) -> SE gamma (k2) -> 1x1 conv WMMA +
// combine (k3) -> 3x3 conv WMMA x2 (conv1 64->128, conv2 128->128).
// GEMM work: v_wmma_f32_16x16x32_bf16 (wave32). Conv halo staging: TDM
// (tensor_load_to_lds) when available, else async-to-LDS, else sync loads.

typedef __attribute__((ext_vector_type(16))) __bf16 v16bf;
typedef __attribute__((ext_vector_type(8)))  __bf16 v8bf;
typedef __attribute__((ext_vector_type(8)))  float  v8f;
typedef __attribute__((ext_vector_type(4)))  unsigned int v4u;
typedef __attribute__((ext_vector_type(8)))  int v8i;
typedef __attribute__((ext_vector_type(4)))  int v4i;

#define BB   8
#define CI1  64
#define CO   128
#define WW   384     // raw input W (pre-pool)
#define HP   192     // pooled H
#define WP   192     // pooled W

#if __has_builtin(__builtin_amdgcn_tensor_load_to_lds)
#define HAVE_TDM 1
#else
#define HAVE_TDM 0
#endif
#if __has_builtin(__builtin_amdgcn_global_load_async_to_lds_b128)
#define HAVE_ASYNC 1
#else
#define HAVE_ASYNC 0
#endif

// cos(pi*(t+0.5)/8) and cos(pi*(t+0.5)*2/8)
__device__ __constant__ float C1T[8] = {
    0.98078528f, 0.83146961f, 0.55557023f, 0.19509032f,
   -0.19509032f,-0.55557023f,-0.83146961f,-0.98078528f};
__device__ __constant__ float C2T[8] = {
    0.92387953f, 0.38268343f,-0.38268343f,-0.92387953f,
   -0.92387953f,-0.38268343f, 0.38268343f, 0.92387953f};

#if HAVE_TDM
// Issue one 1-D TDM copy: nelem bf16 elements from gsrc -> LDS at lds_addr,
// inserting pad_amount(code 3 -> 4 DWORDs = 16B) after every
// 2^(pad_code) 8-byte units (code 4 -> 128B, code 5 -> 256B): matches the
// PITCH = CI+8 bank-stagger layout of the halo tile. ISA 8.3-8.6.
__device__ inline void tdm_load_row(const void* gsrc, unsigned lds_addr,
                                    unsigned nelem, unsigned pad_code) {
  unsigned long long ga = (unsigned long long)(size_t)gsrc;
  v4u g0;
  g0.x = 1u;                                    // count=1 (valid user desc)
  g0.y = lds_addr;                              // [63:32] lds_addr
  g0.z = (unsigned)(ga & 0xffffffffu);          // [95:64] global_addr lo
  g0.w = (unsigned)((ga >> 32) & 0x01ffffffu)   // [120:96] global_addr hi
         | (2u << 30);                          // [127:126] type=2 (image)
  v8i g1;
  g1[0] = (int)((1u << 16)                      // data_size code 1 = 2B
                | (1u << 20)                    // pad_enable
                | (pad_code << 22)              // pad_interval
                | (3u << 25));                  // pad_amount = 4 DWORDs
  g1[1] = (int)((nelem & 0xffffu) << 16);       // tensor_dim0[15:0] @bit48
  g1[2] = (int)(((nelem >> 16) & 0xffffu)       // tensor_dim0[31:16]
                | (1u << 16));                  // tensor_dim1 = 1
  g1[3] = (int)((nelem & 0xffffu) << 16);       // tile_dim0 @bit112
  g1[4] = 0;                                    // tile_dim1/2 = 0 (1-D tile)
  g1[5] = (int)nelem;                           // tensor_dim0_stride lo
  g1[6] = 0;
  g1[7] = 0;
  v4i gz = {0, 0, 0, 0};
#if defined(__clang_major__) && (__clang_major__ >= 23)
  v8i gz8 = {0, 0, 0, 0, 0, 0, 0, 0};
  __builtin_amdgcn_tensor_load_to_lds(g0, g1, gz, gz, gz8, 0);
#else
  __builtin_amdgcn_tensor_load_to_lds(g0, g1, gz, gz, 0);
#endif
}
#endif

// ---------------------------------------------------------------------------
// k0: pack OIHW fp32 weights into WMMA B-fragment order (bf16).
// h = lane/16: K = kc*32 + 8h + e (e<8), kc*32 + 16 + 8h + (e-8) (e>=8); N = lane%16.
// Storage [chunk][nsub][lane][16] -> per-lane contiguous 32B load.
// ---------------------------------------------------------------------------
__global__ void k0_pack(const float* __restrict__ w, __bf16* __restrict__ out,
                        int Cout, int Cin, int KH, int total) {
  int nSub = Cout >> 4;
  int cpt  = Cin >> 5;
  for (int idx = blockIdx.x * blockDim.x + threadIdx.x; idx < total;
       idx += gridDim.x * blockDim.x) {
    int e     = idx & 15;
    int lane  = (idx >> 4) & 31;
    int nsub  = (idx >> 9) % nSub;
    int chunk = idx / (512 * nSub);
    int tap   = chunk / cpt;
    int kc    = chunk % cpt;
    int kl    = ((e < 8) ? e : (e + 8)) + ((lane < 16) ? 0 : 8);
    int c     = kc * 32 + kl;
    int n     = nsub * 16 + (lane & 15);
    int dy    = tap / KH;
    int dx    = tap % KH;
    out[idx]  = (__bf16)w[((size_t)(n * Cin + c) * KH + dy) * KH + dx];
  }
}

// ---------------------------------------------------------------------------
// k1: per (b,c): maxpool into LDS, 4-coeff block DCT, remapped IDCT -> y1,
// y2 = xp - y1, sum/sumsq reduction.
// ---------------------------------------------------------------------------
__global__ void k1_pool_dct(const float* __restrict__ x, float* __restrict__ y1o,
                            float* __restrict__ y2o, float* __restrict__ stats) {
  const int bc  = blockIdx.x;
  const int tid = threadIdx.x;
  extern __shared__ float sm[];
  float* xp   = sm;                  // 36864
  float* coef = sm + HP * WP;        // 2304
  float* red  = coef + 576 * 4;      // 512
  const float* xin = x + (size_t)bc * WW * WW;

  for (int idx = tid; idx < HP * WP; idx += 256) {
    int p = idx / WP, q = idx % WP;
    const float2 a = *(const float2*)(xin + (size_t)(2 * p) * WW + 2 * q);
    const float2 b = *(const float2*)(xin + (size_t)(2 * p + 1) * WW + 2 * q);
    xp[idx] = fmaxf(fmaxf(a.x, a.y), fmaxf(b.x, b.y));
  }
  __syncthreads();

  for (int blk = tid; blk < 576; blk += 256) {
    int bi = blk / 24, bj = blk % 24;
    float a0 = 0.f, a1 = 0.f, a2 = 0.f, a3 = 0.f;
    const float* base = xp + (bi * 8) * WP + bj * 8;
    #pragma unroll
    for (int p = 0; p < 8; ++p) {
      float rc = C1T[p];
      #pragma unroll
      for (int q = 0; q < 8; ++q) {
        float v = base[p * WP + q];
        a0 += v; a1 += v * C1T[q]; a2 += v * rc; a3 += v * C2T[q];
      }
    }
    const float s = 0.17677669529663687f;  // sqrt(2)/8
    coef[blk * 4 + 0] = a0 * 0.125f;
    coef[blk * 4 + 1] = a1 * s;
    coef[blk * 4 + 2] = a2 * s;
    coef[blk * 4 + 3] = a3 * s;
  }
  __syncthreads();

  float sum = 0.f, ssq = 0.f;
  float* y1p = y1o + (size_t)bc * HP * WP;
  float* y2p = y2o + (size_t)bc * HP * WP;
  for (int idx = tid; idx < HP * WP; idx += 256) {
    int p = idx / WP, q = idx % WP;
    int hp = (p * 185) / 192, wq = (q * 185) / 192;
    int bi = hp >> 3, up = hp & 7;
    int bj = wq >> 3, vq = wq & 7;
    const float* c = coef + (bi * 24 + bj) * 4;
    const float s = 0.17677669529663687f;
    float v = c[0] * 0.125f + s * (c[1] * C1T[vq] + c[2] * C1T[up] + c[3] * C2T[vq]);
    y1p[idx] = v;
    y2p[idx] = xp[idx] - v;
    sum += v; ssq += v * v;
  }
  red[tid] = sum;
  red[256 + tid] = ssq;
  __syncthreads();
  for (int s2 = 128; s2 > 0; s2 >>= 1) {
    if (tid < s2) {
      red[tid] += red[tid + s2];
      red[256 + tid] += red[256 + tid + s2];
    }
    __syncthreads();
  }
  if (tid == 0) {
    stats[bc * 2 + 0] = red[0];
    stats[bc * 2 + 1] = red[256];
  }
}

// ---------------------------------------------------------------------------
// k2: SE MLP: s=avg+var; t=relu(fc1*s); gamma=sigmoid(fc2*t).
// ---------------------------------------------------------------------------
__global__ void k2_gamma(const float* __restrict__ stats,
                         const float* __restrict__ fc1,
                         const float* __restrict__ fc2,
                         float* __restrict__ gamma) {
  int b = blockIdx.x;
  int c = threadIdx.x;  // 64 threads
  __shared__ float s[64];
  __shared__ float t[4];
  float sum = stats[(b * 64 + c) * 2 + 0];
  float ssq = stats[(b * 64 + c) * 2 + 1];
  const float n = 36864.f;
  float avg = sum / n;
  float var = (ssq - sum * sum / n) / (n - 1.f);
  s[c] = avg + var;
  __syncthreads();
  if (c < 4) {
    float acc = 0.f;
    for (int k = 0; k < 64; ++k) acc += fc1[c * 64 + k] * s[k];
    t[c] = fmaxf(acc, 0.f);
  }
  __syncthreads();
  float g = 0.f;
  #pragma unroll
  for (int o = 0; o < 4; ++o) g += fc2[c * 4 + o] * t[o];
  gamma[b * 64 + c] = 1.f / (1.f + __expf(-g));
}

// Load one 16x32 bf16 A fragment from an LDS row (ISA 7.12.2 layout).
__device__ inline v16bf load_afrag(const __bf16* ap, int c0) {
  v8bf lo = *reinterpret_cast<const v8bf*>(ap + c0);
  v8bf hi = *reinterpret_cast<const v8bf*>(ap + c0 + 16);
  return __builtin_shufflevector(lo, hi, 0, 1, 2, 3, 4, 5, 6, 7, 8, 9, 10, 11,
                                 12, 13, 14, 15);
}

// ---------------------------------------------------------------------------
// k3: 1x1 conv (N=64,K=64) as WMMA GEMM + combine -> x_all NHWC bf16.
// ---------------------------------------------------------------------------
__global__ void k3_combine(const float* __restrict__ y1,
                           const float* __restrict__ y2,
                           const float* __restrict__ gamma,
                           const __bf16* __restrict__ wpa,
                           const float* __restrict__ attb,
                           __bf16* __restrict__ xall) {
  const int yrow = blockIdx.x % HP;
  const int x0 = (blockIdx.x / HP) * 64;
  const int b = blockIdx.z;
  const int tid = threadIdx.x;
  const int lane = tid & 31, wave = tid >> 5;
  const int wm = wave & 3, wn = wave >> 2;
  constexpr int PITCH = CI1 + 8;  // 72 bf16
  __shared__ __align__(16) __bf16 at[64 * PITCH];

  for (int chunk = tid; chunk < CI1 * 16; chunk += 256) {
    int c = chunk >> 4;
    int xq = chunk & 15;
    const float g = gamma[b * 64 + c];
    const float4 v = *(const float4*)(y1 + (((size_t)(b * 64 + c)) * HP + yrow) * WP + x0 + xq * 4);
    int m = xq * 4;
    at[(m + 0) * PITCH + c] = (__bf16)(v.x * g);
    at[(m + 1) * PITCH + c] = (__bf16)(v.y * g);
    at[(m + 2) * PITCH + c] = (__bf16)(v.z * g);
    at[(m + 3) * PITCH + c] = (__bf16)(v.w * g);
  }
  __syncthreads();

  v8f acc[2] = {};
  const int mrow = lane & 15;
  #pragma unroll
  for (int kc = 0; kc < 2; ++kc) {
    int c0 = kc * 32 + ((lane >= 16) ? 8 : 0);
    const __bf16* ap = at + (wm * 16 + mrow) * PITCH;
    v16bf afrag = load_afrag(ap, c0);
    #pragma unroll
    for (int t = 0; t < 2; ++t) {
      int nsub = wn * 2 + t;
      const v16bf bfrag =
          *reinterpret_cast<const v16bf*>(wpa + (((size_t)kc * 4 + nsub) * 32 + lane) * 16);
      acc[t] = __builtin_amdgcn_wmma_f32_16x16x32_bf16(
          false, afrag, false, bfrag, (short)0, acc[t], false, false);
    }
  }

  const int hi16 = lane >> 4;
  const int xbase = x0 + wm * 16 + hi16 * 8;
  #pragma unroll
  for (int t = 0; t < 2; ++t) {
    int n = wn * 32 + t * 16 + (lane & 15);
    float bv = attb[n];
    const float* y2p = y2 + (((size_t)(b * 64 + n)) * HP + yrow) * WP + xbase;
    __bf16* op = xall + (((size_t)b * HP + yrow) * WP + xbase) * CI1 + n;
    #pragma unroll
    for (int r = 0; r < 8; ++r) {
      float v = fmaxf(acc[t][r] + bv, 0.f) + y2p[r];
      op[r * CI1] = (__bf16)v;
    }
  }
}

// ---------------------------------------------------------------------------
// 3x3 conv via implicit GEMM. Input NHWC bf16, M=64-pixel row tile, N=128.
// Halo tile 3x66xCI staged in LDS by TDM / async-to-LDS / sync loads.
// ---------------------------------------------------------------------------
template <int CI, bool FINAL>
__global__ void conv3x3_wmma(const __bf16* __restrict__ in,
                             const __bf16* __restrict__ wp,
                             const float* __restrict__ bias,
                             __bf16* __restrict__ outb,
                             float* __restrict__ outf) {
  const int yrow = blockIdx.x % HP;
  const int x0 = (blockIdx.x / HP) * 64;
  const int b = blockIdx.z;
  const int tid = threadIdx.x;
  const int lane = tid & 31, wave = tid >> 5;
  const int wm = wave & 3;
  const int wn = wave >> 2;
  constexpr int PITCH = CI + 8;
  // +16 slack elements: TDM trailing pad write past the last pixel
  __shared__ __align__(16) __bf16 tile[3 * 66 * PITCH + 16];

  constexpr int CPP = CI / 8;   // 16B chunks per pixel
  constexpr int NCH = 3 * 66 * CPP;

#if HAVE_TDM
  if ((tid >> 5) == 0) {  // one wave issues the DMAs (TDM ignores EXEC)
    #pragma unroll
    for (int r = 0; r < 3; ++r) {
      int gy = yrow - 1 + r;
      if (gy >= 0 && gy < HP) {
        int gxs = (x0 - 1 < 0) ? 0 : (x0 - 1);
        int gxe = (x0 + 64 > WP - 1) ? (WP - 1) : (x0 + 64);
        unsigned npix = (unsigned)(gxe - gxs + 1);
        const __bf16* gsrc = in + (((size_t)b * HP + gy) * WP + gxs) * CI;
        unsigned lds = (unsigned)(size_t)(const void*)(
            tile + ((size_t)(r * 66 + (gxs - (x0 - 1)))) * PITCH);
        tdm_load_row(gsrc, lds, npix * CI, (CI == 64) ? 4u : 5u);
      }
    }
  }
  __builtin_amdgcn_s_wait_tensorcnt(0);
  __syncthreads();
  // zero-fill the halo pixels the DMAs did not cover (image border)
  for (int ch = tid; ch < NCH; ch += 256) {
    int pix = ch / CPP, sub = ch % CPP;
    int r = pix / 66, xl = pix % 66;
    int gy = yrow - 1 + r, gx = x0 - 1 + xl;
    if (!(gy >= 0 && gy < HP && gx >= 0 && gx < WP))
      *reinterpret_cast<uint4*>(tile + ((size_t)(r * 66 + xl)) * PITCH + sub * 8) =
          uint4{0u, 0u, 0u, 0u};
  }
  __syncthreads();
#else
  for (int ch = tid; ch < NCH; ch += 256) {
    int pix = ch / CPP, sub = ch % CPP;
    int r = pix / 66, xl = pix % 66;
    int gy = yrow - 1 + r, gx = x0 - 1 + xl;
    __bf16* ldst = tile + ((size_t)(r * 66 + xl)) * PITCH + sub * 8;
    bool inr = (gy >= 0 && gy < HP && gx >= 0 && gx < WP);
#if HAVE_ASYNC
    if (inr) {
      __builtin_amdgcn_global_load_async_to_lds_b128(
          (__attribute__((address_space(1))) void*)(
              in + (((size_t)b * HP + gy) * WP + gx) * CI + sub * 8),
          (__attribute__((address_space(3))) void*)ldst, 0, 0);
    } else {
      *reinterpret_cast<uint4*>(ldst) = uint4{0u, 0u, 0u, 0u};
    }
#else
    uint4 v{0u, 0u, 0u, 0u};
    if (inr)
      v = *reinterpret_cast<const uint4*>(
          in + (((size_t)b * HP + gy) * WP + gx) * CI + sub * 8);
    *reinterpret_cast<uint4*>(ldst) = v;
#endif
  }
#if HAVE_ASYNC
#if __has_builtin(__builtin_amdgcn_s_wait_asynccnt)
  __builtin_amdgcn_s_wait_asynccnt(0);
#else
  asm volatile("s_wait_asynccnt 0x0" ::: "memory");
#endif
#endif
  __syncthreads();
#endif

  __builtin_prefetch(wp, 0, 1);  // global_prefetch_b8 on the weight slab

  v8f acc[4] = {};
  const int mrow = lane & 15;
  constexpr int CPT = CI / 32;   // K-chunks per tap
  constexpr int NSUB = CO / 16;  // 8

  #pragma unroll
  for (int tap = 0; tap < 9; ++tap) {
    const int dy = tap / 3, dx = tap % 3;
    const __bf16* ap = tile + ((size_t)(dy * 66 + wm * 16 + mrow + dx)) * PITCH;
    #pragma unroll
    for (int kc = 0; kc < CPT; ++kc) {
      int c0 = kc * 32 + ((lane >= 16) ? 8 : 0);
      v16bf afrag = load_afrag(ap, c0);
      int chunk = tap * CPT + kc;
      #pragma unroll
      for (int t = 0; t < 4; ++t) {
        int nsub = wn * 4 + t;
        const v16bf bfrag = *reinterpret_cast<const v16bf*>(
            wp + (((size_t)chunk * NSUB + nsub) * 32 + lane) * 16);
        acc[t] = __builtin_amdgcn_wmma_f32_16x16x32_bf16(
            false, afrag, false, bfrag, (short)0, acc[t], false, false);
      }
    }
  }

  const int hi16 = lane >> 4;
  const int xbase = x0 + wm * 16 + hi16 * 8;
  #pragma unroll
  for (int t = 0; t < 4; ++t) {
    int n = wn * 64 + t * 16 + (lane & 15);
    float bv = bias[n];
    if constexpr (FINAL) {
      float v0 = fmaxf(acc[t][0] + bv, 0.f), v1 = fmaxf(acc[t][1] + bv, 0.f);
      float v2 = fmaxf(acc[t][2] + bv, 0.f), v3 = fmaxf(acc[t][3] + bv, 0.f);
      float v4 = fmaxf(acc[t][4] + bv, 0.f), v5 = fmaxf(acc[t][5] + bv, 0.f);
      float v6 = fmaxf(acc[t][6] + bv, 0.f), v7 = fmaxf(acc[t][7] + bv, 0.f);
      float* op = outf + (((size_t)b * CO + n) * HP + yrow) * WP + xbase;
      *reinterpret_cast<float4*>(op + 0) = make_float4(v0, v1, v2, v3);
      *reinterpret_cast<float4*>(op + 4) = make_float4(v4, v5, v6, v7);
    } else {
      __bf16* op = outb + (((size_t)b * HP + yrow) * WP + xbase) * CO + n;
      #pragma unroll
      for (int r = 0; r < 8; ++r)
        op[r * CO] = (__bf16)fmaxf(acc[t][r] + bv, 0.f);
    }
  }
}

// ---------------------------------------------------------------------------
extern "C" void kernel_launch(void* const* d_in, const int* in_sizes, int n_in,
                              void* d_out, int out_size, void* d_ws,
                              size_t ws_size, hipStream_t stream) {
  (void)in_sizes; (void)n_in; (void)out_size; (void)ws_size;
  const float* x    = (const float*)d_in[0];
  const float* w1   = (const float*)d_in[1];
  const float* b1   = (const float*)d_in[2];
  const float* w2   = (const float*)d_in[3];
  const float* b2   = (const float*)d_in[4];
  const float* attw = (const float*)d_in[5];
  const float* attb = (const float*)d_in[6];
  const float* fc1  = (const float*)d_in[7];
  const float* fc2  = (const float*)d_in[8];

  char* ws = (char*)d_ws;
  size_t off = 0;
  auto carve = [&](size_t bytes) -> char* {
    char* p = ws + off;
    off = (off + bytes + 255) & ~(size_t)255;
    return p;
  };
  float*  y1    = (float*)carve((size_t)BB * 64 * HP * WP * 4);
  float*  y2    = (float*)carve((size_t)BB * 64 * HP * WP * 4);
  __bf16* xall  = (__bf16*)carve((size_t)BB * HP * WP * 64 * 2);
  __bf16* mid   = (__bf16*)carve((size_t)BB * HP * WP * 128 * 2);
  __bf16* wp1   = (__bf16*)carve(73728 * 2);
  __bf16* wp2   = (__bf16*)carve(147456 * 2);
  __bf16* wpa   = (__bf16*)carve(4096 * 2);
  float*  stats = (float*)carve(512 * 2 * 4);
  float*  gam   = (float*)carve(512 * 4);

  k0_pack<<<(73728 + 255) / 256, 256, 0, stream>>>(w1, wp1, 128, 64, 3, 73728);
  k0_pack<<<(147456 + 255) / 256, 256, 0, stream>>>(w2, wp2, 128, 128, 3, 147456);
  k0_pack<<<(4096 + 255) / 256, 256, 0, stream>>>(attw, wpa, 64, 64, 1, 4096);

  k1_pool_dct<<<512, 256, (36864 + 2304 + 512) * 4, stream>>>(x, y1, y2, stats);
  k2_gamma<<<8, 64, 0, stream>>>(stats, fc1, fc2, gam);
  k3_combine<<<dim3(3 * HP, 1, BB), 256, 0, stream>>>(y1, y2, gam, wpa, attb, xall);

  conv3x3_wmma<64, false><<<dim3(3 * HP, 1, BB), 256, 0, stream>>>(
      xall, wp1, b1, mid, nullptr);
  conv3x3_wmma<128, true><<<dim3(3 * HP, 1, BB), 256, 0, stream>>>(
      mid, wp2, b2, nullptr, (float*)d_out);
}
